// SimFocusChamferLoss2D_41824391529115
// MI455X (gfx1250) — matrix operand order repaired and checked by reading, not compiled, not run on hardware
//
#include <hip/hip_runtime.h>
#include <math.h>

// ---------------- problem constants ----------------
#define N_OBJ 16
#define N_P   128
#define N2    2048
#define CCH   256
#define FH    384
#define FW    384
#define IMG_W 384.0f
#define IMG_H 384.0f
#define NPTS1 (N_OBJ * N_P)      // 2048 key points
#define NPTS2 (N_OBJ * N2)       // 32768 candidate points
#define NPTS  (NPTS1 + NPTS2)    // 34816

typedef __attribute__((ext_vector_type(16))) __bf16 v16bf;
typedef __attribute__((ext_vector_type(8)))  __bf16 v8bf;
typedef __attribute__((ext_vector_type(8)))  float  v8f;

union Frag {
    v16bf v;
    struct { v8bf lo, hi; } h;
};

// ---------------------------------------------------------------------------
// CDNA5 async global->LDS copy helpers (ASYNCcnt-tracked, see CDNA5 ISA)
// ---------------------------------------------------------------------------
__device__ __forceinline__ void async_copy_32B(const void* gsrc, void* lds_dst) {
    // VDST = LDS byte offset (low 32 bits of the flat shared pointer),
    // VADDR = 64-bit global address. INST_OFFSET is added to both sides.
    unsigned           loff = (unsigned)(size_t)lds_dst;
    unsigned long long ga   = (unsigned long long)(size_t)gsrc;
    asm volatile(
        "global_load_async_to_lds_b128 %0, %1, off\n"
        "global_load_async_to_lds_b128 %0, %1, off offset:16\n"
        :: "v"(loff), "v"(ga) : "memory");
}

__device__ __forceinline__ void wait_async0() {
    asm volatile("s_wait_asynccnt 0x0" ::: "memory");
}

// ---------------------------------------------------------------------------
// Kernel 1: bilinear grid-sample (align_corners=False, zero pad) -> bf16 rows
// block = 256 threads (one per channel), grid = 34816 points
// ---------------------------------------------------------------------------
__global__ __launch_bounds__(256) void sample_kernel(
    const float* __restrict__ feats,      // [C,H,W]
    const float* __restrict__ key_points, // [2048,2]
    const float* __restrict__ ps2,        // [32768,2] flattened
    __bf16* __restrict__ f1b,             // [2048,256]
    __bf16* __restrict__ f2b)             // [32768,256]
{
    const int pt = blockIdx.x;
    const int c  = threadIdx.x;

    float px, py;
    __bf16* outp;
    if (pt < NPTS1) {
        px = key_points[pt * 2 + 0];
        py = key_points[pt * 2 + 1];
        outp = f1b + (size_t)pt * CCH;
    } else {
        const int q = pt - NPTS1;
        px = ps2[q * 2 + 0];
        py = ps2[q * 2 + 1];
        outp = f2b + (size_t)q * CCH;
    }

    // x = p01*W - 0.5 (grid_sample align_corners=False)
    const float x = px * ((float)FW / IMG_W) - 0.5f;
    const float y = py * ((float)FH / IMG_H) - 0.5f;
    const float x0f = floorf(x), y0f = floorf(y);
    const int x0 = (int)x0f, y0 = (int)y0f;
    const float wx1 = x - x0f, wy1 = y - y0f;
    const float wx0 = 1.0f - wx1, wy0 = 1.0f - wy1;

    const float* plane = feats + (size_t)c * (FH * FW);

    float acc = 0.0f;
    {
        int xi = x0, yi = y0;
        if (xi >= 0 && xi < FW && yi >= 0 && yi < FH) acc += plane[yi * FW + xi] * (wx0 * wy0);
    }
    {
        int xi = x0 + 1, yi = y0;
        if (xi >= 0 && xi < FW && yi >= 0 && yi < FH) acc += plane[yi * FW + xi] * (wx1 * wy0);
    }
    {
        int xi = x0, yi = y0 + 1;
        if (xi >= 0 && xi < FW && yi >= 0 && yi < FH) acc += plane[yi * FW + xi] * (wx0 * wy1);
    }
    {
        int xi = x0 + 1, yi = y0 + 1;
        if (xi >= 0 && xi < FW && yi >= 0 && yi < FH) acc += plane[yi * FW + xi] * (wx1 * wy1);
    }

    outp[c] = (__bf16)acc;
}

// ---------------------------------------------------------------------------
// Kernel 2: init reduction buffers
// ---------------------------------------------------------------------------
__global__ void init_kernel(unsigned* __restrict__ dmin_bits,
                            float* __restrict__ dsum,
                            float* __restrict__ dcnt)
{
    const int i = blockIdx.x * blockDim.x + threadIdx.x;
    if (i < NPTS1) {
        dmin_bits[i] = 0x7F800000u;  // +inf
        dsum[i] = 0.0f;
        dcnt[i] = 0.0f;
    }
}

// ---------------------------------------------------------------------------
// Kernel 3: fused WMMA GEMM (sign of f1.f2) + masked distance min/sum/count.
// grid = 16 objects * 8 N-slices; block = 256 threads (8 waves).
// B tiles (16 cols x 256 K bf16 = 8 KB) are double-buffered in LDS via
// global_load_async_to_lds_b128; each byte fetched once per block (vs 8x).
// Wave w computes M rows [16w,16w+16): per tile 16 ds_load_b128 + 8 WMMAs.
// ---------------------------------------------------------------------------
__global__ __launch_bounds__(256) void gemm_mask_kernel(
    const float*  __restrict__ ps1,        // [16,128,2]
    const float*  __restrict__ ps2,        // [16,2048,2]
    const __bf16* __restrict__ f1b,        // [2048,256]
    const __bf16* __restrict__ f2b,        // [32768,256]
    unsigned* __restrict__ dmin_bits,      // [2048]
    float* __restrict__ dsum,              // [2048]
    float* __restrict__ dcnt)              // [2048]
{
    __shared__ __align__(16) __bf16 Bsh[2][16 * CCH];   // 2 x 8 KB

    const int o     = blockIdx.x >> 3;   // object
    const int slice = blockIdx.x & 7;    // 256-wide N slice
    const int wave  = threadIdx.x >> 5;  // 8 waves -> M tile
    const int lane  = threadIdx.x & 31;
    const int hi    = lane >> 4;         // lane half
    const int lm    = lane & 15;
    const int tid   = threadIdx.x;

    const __bf16* Bg = f2b + ((size_t)(o * N2 + slice * 256)) * CCH;

    // ---- kick off async copy of tile 0 while we set up A ----
    async_copy_32B(Bg + tid * 16, &Bsh[0][tid * 16]);

    // ---- preload A fragments (16 rows x 256 K, 8 chunks of K=32) ----
    // A 16x32 bf16 layout: lanes 0-15 hold K [k0..k0+15],
    //                      lanes 16-31 hold K [k0+16..k0+31].
    const __bf16* Abase = f1b + ((size_t)(o * N_P + wave * 16 + lm)) * CCH + hi * 16;
    Frag afrag[8];
#pragma unroll
    for (int kc = 0; kc < 8; ++kc) {
        const __bf16* p = Abase + kc * 32;
        afrag[kc].h.lo = *(const v8bf*)(p);
        afrag[kc].h.hi = *(const v8bf*)(p + 8);
    }

    // ---- this lane's 8 accumulator rows: M = r + 8*hi ----
    float x1[8], y1[8];
#pragma unroll
    for (int r = 0; r < 8; ++r) {
        const int m = wave * 16 + r + 8 * hi;
        x1[r] = ps1[(o * N_P + m) * 2 + 0];
        y1[r] = ps1[(o * N_P + m) * 2 + 1];
    }

    const float inf = __builtin_inff();
    float vmin[8], vsum[8], vcnt[8];
#pragma unroll
    for (int r = 0; r < 8; ++r) { vmin[r] = inf; vsum[r] = 0.0f; vcnt[r] = 0.0f; }

    // ---- sweep 16 N-tiles of this slice, double-buffered through LDS ----
    for (int t = 0; t < 16; ++t) {
        const int cur = t & 1;

        wait_async0();       // my tile-t async writes have landed in LDS
        __syncthreads();     // everyone's copies landed; prev-tile reads done

        if (t + 1 < 16)      // overlap next copy with this tile's compute
            async_copy_32B(Bg + (size_t)(t + 1) * 16 * CCH + tid * 16,
                           &Bsh[cur ^ 1][tid * 16]);

        // B 32x16 bf16 layout: lane holds column n=lm, K [k0+16*hi .. +16)
        const __bf16* shp = &Bsh[cur][lm * CCH + hi * 16];
        Frag bfr[8];
#pragma unroll
        for (int kc = 0; kc < 8; ++kc) {
            bfr[kc].h.lo = *(const v8bf*)(shp + kc * 32);
            bfr[kc].h.hi = *(const v8bf*)(shp + kc * 32 + 8);
        }

        v8f acc = {};
#pragma unroll
        for (int kc = 0; kc < 8; ++kc) {
            acc = __builtin_amdgcn_wmma_f32_16x16x32_bf16(
                false, afrag[kc].v, false, bfr[kc].v,
                (short)0, acc, false, false);
        }

        // geometric distance for this lane's column
        const int n = slice * 256 + t * 16 + lm;
        const float x2 = ps2[(o * N2 + n) * 2 + 0];
        const float y2 = ps2[(o * N2 + n) * 2 + 1];
#pragma unroll
        for (int r = 0; r < 8; ++r) {
            const float dx = x1[r] - x2;
            const float dy = y1[r] - y2;
            const float d  = sqrtf(dx * dx + dy * dy);
            const bool  mk = acc[r] >= 0.0f;   // sign(dot) == sign(cosine sim)
            vmin[r] = (mk && d < vmin[r]) ? d : vmin[r];
            vsum[r] += mk ? d : 0.0f;
            vcnt[r] += mk ? 1.0f : 0.0f;
        }
    }

    // ---- reduce over the 16 lanes of each half (N varies with lm) ----
#pragma unroll
    for (int off = 1; off < 16; off <<= 1) {
#pragma unroll
        for (int r = 0; r < 8; ++r) {
            vmin[r] = fminf(vmin[r], __shfl_xor(vmin[r], off, 32));
            vsum[r] += __shfl_xor(vsum[r], off, 32);
            vcnt[r] += __shfl_xor(vcnt[r], off, 32);
        }
    }

    if (lm == 0) {   // lanes 0 (M=r) and 16 (M=8+r) publish
#pragma unroll
        for (int r = 0; r < 8; ++r) {
            const int p = o * N_P + wave * 16 + r + 8 * hi;
            atomicMin(&dmin_bits[p], __float_as_uint(vmin[r]));  // d >= 0: bit order == float order
            atomicAdd(&dsum[p], vsum[r]);
            atomicAdd(&dcnt[p], vcnt[r]);
        }
    }
}

// ---------------------------------------------------------------------------
// Kernel 4: finalize -> scalar loss
// ---------------------------------------------------------------------------
__global__ __launch_bounds__(256) void final_kernel(
    const float* __restrict__ ps2,
    const unsigned* __restrict__ dmin_bits,
    const float* __restrict__ dsum,
    const float* __restrict__ dcnt,
    float* __restrict__ out)
{
    __shared__ float mfac[N_OBJ];
    __shared__ float red[256];
    const int t = threadIdx.x;

    if (t < N_OBJ) {
        float s = 0.0f;
        const float* p = ps2 + (size_t)t * N2 * 2;
        for (int i = 0; i < N2 * 2; ++i) s += p[i];
        mfac[t] = (s >= 0.0f) ? 1.0f : 0.0f;
    }
    __syncthreads();

    float acc = 0.0f;
    for (int p = t; p < NPTS1; p += 256) {
        const float cnt = dcnt[p];
        float cost = 0.0f;
        if (cnt > 0.0f) {
            const float mn = __uint_as_float(dmin_bits[p]);
            cost = 0.5f * (mn + dsum[p] / cnt);
        }
        acc += cost * mfac[p >> 7];
    }
    red[t] = acc;
    __syncthreads();
    for (int s = 128; s > 0; s >>= 1) {
        if (t < s) red[t] += red[t + s];
        __syncthreads();
    }
    if (t == 0) out[0] = red[0] * (1.0f / (float)(N_P * N_OBJ));
}

// ---------------------------------------------------------------------------
extern "C" void kernel_launch(void* const* d_in, const int* in_sizes, int n_in,
                              void* d_out, int out_size, void* d_ws, size_t ws_size,
                              hipStream_t stream) {
    (void)in_sizes; (void)n_in; (void)out_size; (void)ws_size;
    const float* ps1   = (const float*)d_in[0];  // [16,128,2]
    const float* ps2   = (const float*)d_in[1];  // [16,2048,2]
    const float* feats = (const float*)d_in[2];  // [1,256,384,384]
    const float* kp    = (const float*)d_in[3];  // [2048,2]
    float* out = (float*)d_out;

    char* ws = (char*)d_ws;
    __bf16*   f1b  = (__bf16*)ws;                                  //  1 MB
    __bf16*   f2b  = (__bf16*)(ws + (size_t)NPTS1 * CCH * 2);      // 16 MB
    unsigned* dmin = (unsigned*)(ws + (size_t)(NPTS1 + NPTS2) * CCH * 2);
    float*    dsum = (float*)((char*)dmin + NPTS1 * sizeof(unsigned));
    float*    dcnt = (float*)((char*)dsum + NPTS1 * sizeof(float));

    init_kernel<<<(NPTS1 + 255) / 256, 256, 0, stream>>>(dmin, dsum, dcnt);
    sample_kernel<<<NPTS, 256, 0, stream>>>(feats, kp, ps2, f1b, f2b);
    gemm_mask_kernel<<<N_OBJ * 8, 256, 0, stream>>>(ps1, ps2, f1b, f2b, dmin, dsum, dcnt);
    final_kernel<<<1, 256, 0, stream>>>(ps2, dmin, dsum, dcnt, out);
}
